// THWaC4_59304908423197
// MI455X (gfx1250) — compile-verified, matrix operand order checked by reading, not run on hardware
//
#include <hip/hip_runtime.h>
#include <hip/hip_bf16.h>
#include <stdint.h>

typedef __attribute__((ext_vector_type(16))) _Float16 v16h;
typedef __attribute__((ext_vector_type(8)))  float    v8f;

// ---------------- problem constants ----------------
namespace thw {
constexpr int B = 8, T = 8, C = 128, H = 56, W = 56;
constexpr int HW = H * W;            // 3136
constexpr int HW4 = HW / 4;          // 784
constexpr int BT = B * T;            // 64
constexpr float EPS = 1e-5f;

// ---------------- workspace layout (floats, all 16B-aligned sizes) ----------------
constexpr long PT_OFF     = 0;                                    // (B,2,C,H,W)  pool over t
constexpr long PH_OFF     = PT_OFF     + (long)B * 2 * C * HW;    // (B,2,T,C,W)  pool over h
constexpr long PW_OFF     = PH_OFF     + (long)B * 2 * T * C * W; // (B,2,T,H,C)  pool over w
constexpr long PCMAX_OFF  = PW_OFF     + (long)B * 2 * T * H * C; // (BT,HW)      max over c
constexpr long PCMEAN_OFF = PCMAX_OFF  + (long)BT * HW;           // (BT,HW)      mean over c
constexpr long CT_OFF     = PCMEAN_OFF + (long)BT * HW;           // (B,C,H,W)    conv T out
constexpr long CH_OFF     = CT_OFF     + (long)B * C * HW;        // (B,T,C,W)    conv H out
constexpr long CW_OFF     = CH_OFF     + (long)B * T * C * W;     // (B,T,H,C)    conv W out
constexpr long CIN_OFF    = CW_OFF     + (long)B * T * H * C;     // (B,2,T,H,W)  C-branch conv in
constexpr long CC_OFF     = CIN_OFF    + (long)B * 2 * T * HW;    // (B,T,H,W)    conv C out
constexpr long YT_OFF     = CC_OFF     + (long)BT * HW;           // (B,HW)
constexpr long YH_OFF     = YT_OFF     + (long)B * HW;            // (BT,W)
constexpr long YW_OFF     = YH_OFF     + (long)BT * W;            // (BT,H)
constexpr long ATT_OFF    = YW_OFF     + (long)BT * H;            // (BT,HW)
constexpr long YC1_OFF    = ATT_OFF    + (long)BT * HW;           // (BT,HW)
constexpr long ST_OFF     = YC1_OFF    + (long)BT * HW;           // 4 branches x {sum,sumsq,mu,rsig}
constexpr long PART_OFF   = ST_OFF + 32;                          // 256 x {sum,sumsq} partials
} // namespace thw

using namespace thw;

// ============================================================
// Fused t/h/w pools: one pass over x.
// Block = (b,c); per t: stage the 56x56 slab in LDS (b128 traffic),
// reduce rows (pool over w), columns (pool over h), and keep the
// t-reduction (max/sum) in float4 registers.
// ============================================================
__global__ void thw_pool_thw(const float* __restrict__ x, float* __restrict__ pt,
                             float* __restrict__ ph, float* __restrict__ pw) {
  __shared__ __align__(16) float sl[HW];
  const int tid = threadIdx.x;                 // 256
  const int c = blockIdx.x % C;
  const int b = blockIdx.x / C;

  float4 mx[4], sm[4];
#pragma unroll
  for (int k = 0; k < 4; ++k) {
    mx[k] = make_float4(-INFINITY, -INFINITY, -INFINITY, -INFINITY);
    sm[k] = make_float4(0.f, 0.f, 0.f, 0.f);
  }

  for (int t = 0; t < T; ++t) {
    const float4* xs = (const float4*)(x + ((long)(b * T + t) * C + c) * HW);
#pragma unroll
    for (int k = 0; k < 4; ++k) {
      int q = tid + 256 * k;
      if (q < HW4) {
        float4 v = xs[q];
        ((float4*)sl)[q] = v;
        mx[k].x = fmaxf(mx[k].x, v.x); sm[k].x += v.x;
        mx[k].y = fmaxf(mx[k].y, v.y); sm[k].y += v.y;
        mx[k].z = fmaxf(mx[k].z, v.z); sm[k].z += v.z;
        mx[k].w = fmaxf(mx[k].w, v.w); sm[k].w += v.w;
      }
    }
    __syncthreads();
    if (tid < H) {                             // pool over w: row h = tid
      const float* row = sl + tid * W;
      float m = -INFINITY, s = 0.f;
      for (int w = 0; w < W; ++w) { float v = row[w]; m = fmaxf(m, v); s += v; }
      long o = (((long)(b * 2) * T + t) * H + tid) * C + c;
      pw[o] = m;
      pw[o + (long)T * H * C] = s * (1.f / W);
    } else if (tid >= 64 && tid < 64 + W) {    // pool over h: col w = tid-64
      int w = tid - 64;
      float m = -INFINITY, s = 0.f;
      for (int h = 0; h < H; ++h) { float v = sl[h * W + w]; m = fmaxf(m, v); s += v; }
      long o = (((long)(b * 2) * T + t) * C + c) * W + w;
      ph[o] = m;
      ph[o + (long)T * C * W] = s * (1.f / H);
    }
    __syncthreads();
  }

  float4* po0 = (float4*)(pt + ((long)(b * 2 + 0) * C + c) * HW);
  float4* po1 = (float4*)(pt + ((long)(b * 2 + 1) * C + c) * HW);
#pragma unroll
  for (int k = 0; k < 4; ++k) {
    int q = tid + 256 * k;
    if (q < HW4) {
      po0[q] = mx[k];
      float4 s = sm[k];
      s.x *= (1.f / T); s.y *= (1.f / T); s.z *= (1.f / T); s.w *= (1.f / T);
      po1[q] = s;
    }
  }
}

// ============================================================
// Pool over c (vectorized b128)
// ============================================================
__global__ void thw_pool_c(const float* __restrict__ x,
                           float* __restrict__ pcmax, float* __restrict__ pcmean) {
  int i = blockIdx.x * blockDim.x + threadIdx.x;          // (bt, hw/4)
  if (i >= BT * HW4) return;
  int bt = i / HW4, q = i - bt * HW4;
  const float4* px = (const float4*)(x + (long)bt * C * HW) + q;
  float4 mx = make_float4(-INFINITY, -INFINITY, -INFINITY, -INFINITY);
  float4 sm = make_float4(0.f, 0.f, 0.f, 0.f);
  for (int c = 0; c < C; ++c) {
    float4 v = px[(long)c * HW4];
    mx.x = fmaxf(mx.x, v.x); sm.x += v.x;
    mx.y = fmaxf(mx.y, v.y); sm.y += v.y;
    mx.z = fmaxf(mx.z, v.z); sm.z += v.z;
    mx.w = fmaxf(mx.w, v.w); sm.w += v.w;
  }
  ((float4*)pcmax)[i] = mx;
  sm.x *= (1.f / C); sm.y *= (1.f / C); sm.z *= (1.f / C); sm.w *= (1.f / C);
  ((float4*)pcmean)[i] = sm;
}

// ============================================================
// 3x3x3 conv (2 in-ch -> 1 out-ch, SAME) as WMMA implicit GEMM.
// M = 16 consecutive W-positions, K = 54 weights padded to 64
// (two v_wmma_f32_16x16x32_f16), B operand = weights replicated
// across all 16 N columns, so every D column equals the conv output.
// ============================================================
__device__ __forceinline__ int thw_kmap(int j, int hf) {
  int v = j >> 1, s = j & 1;
  return (v < 4) ? (2 * v + s + 8 * hf) : (16 + 2 * (v - 4) + s + 8 * hf);
}

__device__ __forceinline__ _Float16 thw_patch(const float* __restrict__ in,
                                              int D, int Hd, int Wd,
                                              int n, int d, int hh, int ww, int k) {
  int ci = k / 27; int r = k - ci * 27;
  int kd = r / 9; r -= kd * 9;
  int kh = r / 3; int kw = r - kh * 3;
  int id = d + kd - 1, ih = hh + kh - 1, iw = ww + kw - 1;
  if ((unsigned)id >= (unsigned)D || (unsigned)ih >= (unsigned)Hd || (unsigned)iw >= (unsigned)Wd)
    return (_Float16)0.f;
  return (_Float16)in[(((long)(n * 2 + ci) * D + id) * Hd + ih) * Wd + iw];
}

__global__ void thw_conv3d_wmma(const float* __restrict__ in,
                                const float* __restrict__ wgt,
                                float* __restrict__ out,
                                int N, int D, int Hd, int Wd, int n_tiles) {
  const int wave = threadIdx.x >> 5;
  const int lane = threadIdx.x & 31;
  const int tile = blockIdx.x * (blockDim.x >> 5) + wave;
  if (tile >= n_tiles) return;                    // wave-uniform

  const int tpr = (Wd + 15) >> 4;
  const int row = tile / tpr;                     // (n, d, hh)
  const int w0  = (tile - row * tpr) << 4;
  const int hh  = row % Hd;
  const int nd  = row / Hd;
  const int d   = nd % D;
  const int n   = nd / D;

  const int hf = lane >> 4;
  const int m  = lane & 15;                       // A-matrix row (output pixel)
  const int xw = w0 + m;

  v16h a0, a1, b0, b1;
#pragma unroll
  for (int j = 0; j < 16; ++j) {
    int k0 = thw_kmap(j, hf);                     // 0..31
    int k1 = k0 + 32;                             // 32..63 (>=54 zero-padded)
    b0[j] = (_Float16)wgt[k0];
    b1[j] = (k1 < 54) ? (_Float16)wgt[k1] : (_Float16)0.f;
    a0[j] = thw_patch(in, D, Hd, Wd, n, d, hh, xw, k0);
    a1[j] = (k1 < 54) ? thw_patch(in, D, Hd, Wd, n, d, hh, xw, k1) : (_Float16)0.f;
  }

  v8f acc = {};
  acc = __builtin_amdgcn_wmma_f32_16x16x32_f16(false, a0, false, b0, (short)0, acc, false, false);
  acc = __builtin_amdgcn_wmma_f32_16x16x32_f16(false, a1, false, b1, (short)0, acc, false, false);

  int wr = -1;
  if (hf == 0 && m < 8)      wr = m;              // lanes 0..7  -> rows 0..7
  else if (hf == 1 && m < 8) wr = 8 + m;          // lanes 16..23 -> rows 8..15
  if (wr >= 0) {
    int rsel = wr & 7;
    float v = 0.f;
#pragma unroll
    for (int q = 0; q < 8; ++q) if (q == rsel) v = acc[q];
    int xo = w0 + wr;
    if (xo < Wd) out[(long)row * Wd + xo] = v;
  }
}

// ============================================================
// Deterministic two-stage BN statistics (no float atomics),
// stage 1 vectorized b128.
// ============================================================
__global__ void thw_stats_partial(const float* __restrict__ in, long n4,
                                  float* __restrict__ part) {
  __shared__ float ss[256], sq[256];
  int tid = threadIdx.x;
  float a = 0.f, b = 0.f;
  const float4* in4 = (const float4*)in;
  for (long i = (long)blockIdx.x * blockDim.x + tid; i < n4; i += (long)gridDim.x * blockDim.x) {
    float4 v = in4[i];
    a += v.x + v.y + v.z + v.w;
    b += v.x * v.x + v.y * v.y + v.z * v.z + v.w * v.w;
  }
  ss[tid] = a; sq[tid] = b; __syncthreads();
  for (int s = 128; s > 0; s >>= 1) {
    if (tid < s) { ss[tid] += ss[tid + s]; sq[tid] += sq[tid + s]; }
    __syncthreads();
  }
  if (tid == 0) { part[2 * blockIdx.x] = ss[0]; part[2 * blockIdx.x + 1] = sq[0]; }
}

__global__ void thw_stats_final(const float* __restrict__ part,
                                float* __restrict__ st, float count) {
  __shared__ float ss[256], sq[256];
  int tid = threadIdx.x;
  ss[tid] = part[2 * tid]; sq[tid] = part[2 * tid + 1]; __syncthreads();
  for (int s = 128; s > 0; s >>= 1) {
    if (tid < s) { ss[tid] += ss[tid + s]; sq[tid] += sq[tid + s]; }
    __syncthreads();
  }
  if (tid == 0) {
    float mu  = ss[0] / count;
    float var = sq[0] / count - mu * mu;
    st[0] = ss[0]; st[1] = sq[0];
    st[2] = mu;    st[3] = rsqrtf(var + EPS);
  }
}

// ============================================================
// BN + sigmoid + mean over c (generic stride form)
// base = (i/inner)*ostride + (i%inner); addr = base + c*cstride
// st points at {mu, rsig}
// ============================================================
__global__ void thw_sigmean(const float* __restrict__ in, const float* __restrict__ st,
                            const float* __restrict__ g, const float* __restrict__ be,
                            float* __restrict__ out,
                            int nout, int inner, int ostride, int cstride) {
  int i = blockIdx.x * blockDim.x + threadIdx.x;
  if (i >= nout) return;
  float gg = g[0] * st[1];
  float bb = be[0] - st[0] * gg;                  // z = v*gg + bb
  long base = (long)(i / inner) * ostride + (i % inner);
  float s = 0.f;
  for (int c = 0; c < C; ++c) {
    float z = in[base + (long)c * cstride] * gg + bb;
    s += 1.f / (1.f + __expf(-z));
  }
  out[i] = s * (1.f / C);
}

// ============================================================
// att = (yt + yh + yw)/3 ; C-branch conv input = att * {maxc, meanc}
// ============================================================
__global__ void thw_att_cin(const float* __restrict__ yt, const float* __restrict__ yh,
                            const float* __restrict__ yw,
                            const float* __restrict__ pcmax, const float* __restrict__ pcmean,
                            float* __restrict__ att, float* __restrict__ cin) {
  int i = blockIdx.x * blockDim.x + threadIdx.x;  // (bt, hw)
  if (i >= BT * HW) return;
  int bt = i / HW, hw = i - bt * HW;
  int b = bt >> 3, t = bt & 7;
  int h = hw / W, w = hw - h * W;
  float a = (yt[b * HW + hw] + yh[bt * W + w] + yw[bt * H + h]) * (1.f / 3.f);
  att[i] = a;
  cin[((long)(b * 2 + 0) * T + t) * HW + hw] = a * pcmax[i];
  cin[((long)(b * 2 + 1) * T + t) * HW + hw] = a * pcmean[i];
}

// ============================================================
// yc1 = sigmoid(BN(conv_c))
// ============================================================
__global__ void thw_yc1(const float* __restrict__ cc, const float* __restrict__ st,
                        const float* __restrict__ g, const float* __restrict__ be,
                        float* __restrict__ yc1) {
  int i = blockIdx.x * blockDim.x + threadIdx.x;
  if (i >= BT * HW) return;
  float gg = g[0] * st[1];
  float bb = be[0] - st[0] * gg;
  float z = cc[i] * gg + bb;
  yc1[i] = 1.f / (1.f + __expf(-z));
}

// ============================================================
// out = x * att * yc1 ; b128 streaming, att/yc1 staged into LDS
// via async b128 global->LDS loads (gfx1250 async path).
// ============================================================
__global__ void thw_final(const float* __restrict__ x,
                          const float* __restrict__ att,
                          const float* __restrict__ yc1,
                          float* __restrict__ out) {
  __shared__ __align__(16) float s_att[1024];
  __shared__ __align__(16) float s_y1[1024];
  const int tid = threadIdx.x;
  const int bt  = blockIdx.y;
  const int hw4 = blockIdx.x * 1024 + tid * 4;
  const int hcl = hw4 < HW ? hw4 : HW - 4;        // clamp (still 16B aligned)

  {
    uint32_t la = (uint32_t)(uintptr_t)(&s_att[tid * 4]);
    uint64_t ga = (uint64_t)(uintptr_t)(&att[(long)bt * HW + hcl]);
    asm volatile("global_load_async_to_lds_b128 %0, %1, off" :: "v"(la), "v"(ga) : "memory");
    uint32_t lb = (uint32_t)(uintptr_t)(&s_y1[tid * 4]);
    uint64_t gb = (uint64_t)(uintptr_t)(&yc1[(long)bt * HW + hcl]);
    asm volatile("global_load_async_to_lds_b128 %0, %1, off" :: "v"(lb), "v"(gb) : "memory");
  }
#if __has_builtin(__builtin_amdgcn_s_wait_asynccnt)
  __builtin_amdgcn_s_wait_asynccnt(0);
#else
  asm volatile("s_wait_asynccnt 0" ::: "memory");
#endif
  if (hw4 >= HW) return;

  float4 a4 = *(const float4*)&s_att[tid * 4];
  float4 y4 = *(const float4*)&s_y1[tid * 4];
  float4 p;
  p.x = a4.x * y4.x; p.y = a4.y * y4.y; p.z = a4.z * y4.z; p.w = a4.w * y4.w;

  const float4* xp = (const float4*)(x + (long)bt * C * HW) + (hw4 >> 2);
  float4*       op = (float4*)(out + (long)bt * C * HW) + (hw4 >> 2);
#pragma unroll 4
  for (int c = 0; c < C; ++c) {
    if (c + 4 < C) __builtin_prefetch(xp + (long)(c + 4) * HW4, 0, 0);
    float4 v = xp[(long)c * HW4];
    v.x *= p.x; v.y *= p.y; v.z *= p.z; v.w *= p.w;
    op[(long)c * HW4] = v;
  }
}

// ============================================================
// Host-side pipeline
// ============================================================
extern "C" void kernel_launch(void* const* d_in, const int* in_sizes, int n_in,
                              void* d_out, int out_size, void* d_ws, size_t ws_size,
                              hipStream_t stream) {
  (void)in_sizes; (void)n_in; (void)out_size; (void)ws_size;

  const float* x  = (const float*)d_in[0];
  const float* Wt = (const float*)d_in[1];
  const float* Wh = (const float*)d_in[2];
  const float* Ww = (const float*)d_in[3];
  const float* Wc = (const float*)d_in[4];
  const float* gt = (const float*)d_in[5];  const float* bt = (const float*)d_in[6];
  const float* gh = (const float*)d_in[7];  const float* bh = (const float*)d_in[8];
  const float* gw = (const float*)d_in[9];  const float* bw = (const float*)d_in[10];
  const float* gc = (const float*)d_in[11]; const float* bc = (const float*)d_in[12];
  float* out = (float*)d_out;

  float* ws   = (float*)d_ws;
  float* PT   = ws + PT_OFF;
  float* PH   = ws + PH_OFF;
  float* PW   = ws + PW_OFF;
  float* PCMX = ws + PCMAX_OFF;
  float* PCMN = ws + PCMEAN_OFF;
  float* CT   = ws + CT_OFF;
  float* CH   = ws + CH_OFF;
  float* CW   = ws + CW_OFF;
  float* CIN  = ws + CIN_OFF;
  float* CC   = ws + CC_OFF;
  float* YT   = ws + YT_OFF;
  float* YH   = ws + YH_OFF;
  float* YWp  = ws + YW_OFF;
  float* ATT  = ws + ATT_OFF;
  float* YC1  = ws + YC1_OFF;
  float* ST   = ws + ST_OFF;
  float* PART = ws + PART_OFF;

  const int TPB = 256;
  auto blocks = [](long n, int tpb) { return (int)((n + tpb - 1) / tpb); };

  // 1) pools: one fused pass for t/h/w, one pass for c
  thw_pool_thw<<<B * C, TPB, 0, stream>>>(x, PT, PH, PW);
  thw_pool_c<<<blocks((long)BT * HW4, TPB), TPB, 0, stream>>>(x, PCMX, PCMN);

  // 2) WMMA convs for T/H/W branches
  auto conv = [&](const float* in, const float* wgt, float* o, int N, int D, int Hd, int Wd) {
    int tiles = N * D * Hd * ((Wd + 15) >> 4);
    thw_conv3d_wmma<<<(tiles + 7) / 8, 256, 0, stream>>>(in, wgt, o, N, D, Hd, Wd, tiles);
  };
  conv(PT, Wt, CT, B, C, H, W);
  conv(PH, Wh, CH, B, T, C, W);
  conv(PW, Ww, CW, B, T, H, C);

  // 3) BN stats (deterministic two-stage) + normalize+sigmoid+mean over c
  auto stats = [&](const float* in, long n, int br) {
    thw_stats_partial<<<256, 256, 0, stream>>>(in, n / 4, PART);
    thw_stats_final<<<1, 256, 0, stream>>>(PART, ST + 4 * br, (float)n);
  };
  stats(CT, (long)B * C * HW, 0);
  stats(CH, (long)B * T * C * W, 1);
  stats(CW, (long)B * T * H * C, 2);

  thw_sigmean<<<blocks((long)B * HW, TPB), TPB, 0, stream>>>(
      CT, ST + 0 * 4 + 2, gt, bt, YT, B * HW, HW, C * HW, HW);
  thw_sigmean<<<blocks((long)BT * W, TPB), TPB, 0, stream>>>(
      CH, ST + 1 * 4 + 2, gh, bh, YH, BT * W, W, C * W, W);
  thw_sigmean<<<blocks((long)BT * H, TPB), TPB, 0, stream>>>(
      CW, ST + 2 * 4 + 2, gw, bw, YWp, BT * H, 1, C, 1);

  // 4) attention + C-branch conv input (att * pooled-x-over-c)
  thw_att_cin<<<blocks((long)BT * HW, TPB), TPB, 0, stream>>>(YT, YH, YWp, PCMX, PCMN, ATT, CIN);

  // 5) C-branch conv + BN + sigmoid
  conv(CIN, Wc, CC, B, T, H, W);
  stats(CC, (long)BT * HW, 3);
  thw_yc1<<<blocks((long)BT * HW, TPB), TPB, 0, stream>>>(CC, ST + 3 * 4 + 2, gc, bc, YC1);

  // 6) out = x * att * yc1 (b128 streaming, async-LDS staged operands)
  dim3 fg((HW + 1023) / 1024, BT);
  thw_final<<<fg, 256, 0, stream>>>(x, ATT, YC1, out);
}
